// AdaptiveWindowAttention_75514114998346
// MI455X (gfx1250) — matrix operand order repaired and checked by reading, not compiled, not run on hardware
//
#include <hip/hip_runtime.h>
#include <hip/hip_bf16.h>

#define E_ 2048
#define S_ 2048
#define H_ 16
#define D_ 128
#define B_ 2
#define SMAXK 320   // max padded key span: W(<=256)+15+31 rounded up to 32

typedef __attribute__((ext_vector_type(16))) __bf16 v16bf;
typedef __attribute__((ext_vector_type(8)))  __bf16 v8bf;
typedef __attribute__((ext_vector_type(8)))  float  v8f;

// ---- workspace layout (bytes) ----
static constexpr size_t OFF_STATS = 0;                       // f32: sum[2], sumsq[2], xmean[4096]; int window @ f32-idx 4100
static constexpr size_t OFF_XBF   = 64 * 1024;               // (B,S,E) bf16            16 MB
static constexpr size_t OFF_WQKV  = OFF_XBF  + 16777216ULL;  // (3E,E)  bf16            25.2 MB
static constexpr size_t OFF_WOUT  = OFF_WQKV + 25165824ULL;  // (E,E)   bf16             8.4 MB
static constexpr size_t OFF_QKV   = OFF_WOUT + 8388608ULL;   // (B,S,3E) bf16           50.3 MB
static constexpr size_t OFF_QR    = OFF_QKV  + 50331648ULL;  // (B,H,S,D) bf16          16.8 MB
static constexpr size_t OFF_KR    = OFF_QR   + 16777216ULL;  // (B,H,S,D) bf16
static constexpr size_t OFF_VT    = OFF_KR   + 16777216ULL;  // (B,H,D,S) bf16
static constexpr size_t OFF_O     = OFF_VT   + 16777216ULL;  // (B,S,E)   bf16

// A-operand fragment load: lane holds row M=lane&15; contiguous k+16h..+15
static __device__ __forceinline__ v16bf load_a_frag(const __bf16* row, int kb, int hi) {
  union { v16bf v; v8bf h[2]; } u;
  u.h[0] = *(const v8bf*)(row + kb + 16 * hi);
  u.h[1] = *(const v8bf*)(row + kb + 16 * hi + 8);
  return u.v;
}

// ---------------------------------------------------------------- utilities
__global__ void init_stats_kernel(float* stats) {
  if (threadIdx.x < 4) stats[threadIdx.x] = 0.0f;
}

__global__ void f32_to_bf16_kernel(const float* __restrict__ in, __bf16* __restrict__ out, size_t n) {
  size_t i = (size_t)blockIdx.x * blockDim.x + threadIdx.x;
  if (i < n) out[i] = (__bf16)in[i];
}

// per-column mean over seq + global sum/sumsq per batch (for var, ddof=1)
__global__ void colstats_kernel(const float* __restrict__ x, float* __restrict__ stats) {
  const int e = blockIdx.x, b = blockIdx.y, tid = threadIdx.x;
  const float* col = x + (size_t)b * S_ * E_ + e;
  float s = 0.f, s2 = 0.f;
  for (int t = tid; t < S_; t += 256) { float v = col[(size_t)t * E_]; s += v; s2 += v * v; }
  __shared__ float r1[256], r2[256];
  r1[tid] = s; r2[tid] = s2; __syncthreads();
  for (int st = 128; st > 0; st >>= 1) {
    if (tid < st) { r1[tid] += r1[tid + st]; r2[tid] += r2[tid + st]; }
    __syncthreads();
  }
  if (tid == 0) {
    stats[4 + b * E_ + e] = r1[0] / (float)S_;
    atomicAdd(&stats[b], r1[0]);
    atomicAdd(&stats[2 + b], r2[0]);
  }
}

// gating MLP -> dynamic window (single block, 512 threads)
__global__ void window_kernel(const float* __restrict__ stats,
                              const float* __restrict__ w_c1,
                              const float* __restrict__ w_c2,
                              int* __restrict__ winp) {
  __shared__ float red[512];
  __shared__ float wf[2];
  const int j = threadIdx.x;
  for (int b = 0; b < B_; ++b) {
    const float* xm = stats + 4 + b * E_;
    const float* w  = w_c1 + (size_t)j * E_;
    float hsum = 0.f;
    for (int k = 0; k < E_; ++k) hsum += xm[k] * w[k];
    float hval = hsum / (1.f + __expf(-hsum));          // silu
    red[j] = hval * w_c2[j];
    __syncthreads();
    for (int st = 256; st > 0; st >>= 1) { if (j < st) red[j] += red[j + st]; __syncthreads(); }
    if (j == 0) {
      float learned = 1.f / (1.f + __expf(-red[0]));
      const float Nt = (float)S_ * (float)E_;
      float mean = stats[b] / Nt;
      float var  = (stats[2 + b] - Nt * mean * mean) / (Nt - 1.f);
      float vn   = 1.f / (1.f + __expf(-(var * 10.f - 5.f)));
      wf[b] = 64.f + 0.5f * (vn + learned) * 192.f;
    }
    __syncthreads();
  }
  if (j == 0) {
    int w = (int)(0.5f * (wf[0] + wf[1]));   // trunc like astype(int32)
    if (w > S_) w = S_;
    if (w < 64) w = 64;
    winp[0] = w;
  }
}

// RoPE on q,k + scatter: Qr,Kr (B,H,S,D), Vt (B,H,D,S)
__global__ void rope_scatter_kernel(const __bf16* __restrict__ qkv,
                                    __bf16* __restrict__ Qr, __bf16* __restrict__ Kr,
                                    __bf16* __restrict__ Vt) {
  const int d = threadIdx.x;          // 0..127
  const int s = blockIdx.x, h = blockIdx.y, b = blockIdx.z;
  const size_t base = ((size_t)b * S_ + s) * (3 * E_);
  const size_t off  = (size_t)h * D_ + d;
  float q = (float)qkv[base + off];
  float k = (float)qkv[base + E_ + off];
  float v = (float)qkv[base + 2 * E_ + off];
  const size_t offp = (size_t)h * D_ + (d ^ 1);
  float qp = (float)qkv[base + offp];
  float kp = (float)qkv[base + E_ + offp];
  float rq = (d & 1) ? qp : -qp;      // rotate_half pairing
  float rk = (d & 1) ? kp : -kp;
  int dd = (d < 64) ? d : d - 64;     // concatenated freq table
  float ang = (float)s * __powf(10000.0f, -(float)dd / 64.0f);
  float c = cosf(ang), sn = sinf(ang);
  const size_t o = (((size_t)b * H_ + h) * S_ + s) * D_ + d;
  Qr[o] = (__bf16)(q * c + rq * sn);
  Kr[o] = (__bf16)(k * c + rk * sn);
  Vt[(((size_t)b * H_ + h) * D_ + d) * S_ + s] = (__bf16)v;
}

// ---------------------------------------------------------------- WMMA GEMM: C = A @ W^T
// A: (M,K) bf16 row-major, W: (N,K) bf16 row-major.
// Wave tile 64x64 (4x4 fragments, 16 WMMAs per K-step), block = 4 waves = 128x128.
// grid = (N/128, M/128), block = 128 threads.
template <bool F32OUT>
__global__ void __launch_bounds__(128)
gemm_bf16_nt_kernel(const __bf16* __restrict__ A, const __bf16* __restrict__ W,
                    void* __restrict__ Cout, int N, int K) {
  const int lane = threadIdx.x & 31;
  const int wave = threadIdx.x >> 5;
  const int m0 = blockIdx.y * 128 + (wave >> 1) * 64;
  const int n0 = blockIdx.x * 128 + (wave & 1) * 64;
  const int lm = lane & 15;
  const int hi = lane >> 4;

  v8f acc[4][4];
#pragma unroll
  for (int i = 0; i < 4; ++i)
#pragma unroll
    for (int j = 0; j < 4; ++j) acc[i][j] = v8f{};

  const __bf16* arow[4];
  const __bf16* wrow[4];
#pragma unroll
  for (int i = 0; i < 4; ++i) {
    arow[i] = A + (size_t)(m0 + i * 16 + lm) * K;
    wrow[i] = W + (size_t)(n0 + i * 16 + lm) * K;
  }

#pragma unroll 1
  for (int kb = 0; kb < K; kb += 32) {
    // warm WGP$/L2 one K-tile ahead (global_prefetch_b8)
    if (kb + 32 < K) {
      __builtin_prefetch(arow[0] + kb + 32, 0, 1);
      __builtin_prefetch(wrow[0] + kb + 32, 0, 1);
    }
    v16bf a[4];
#pragma unroll
    for (int i = 0; i < 4; ++i) a[i] = load_a_frag(arow[i], kb, hi);
#pragma unroll
    for (int j = 0; j < 4; ++j) {
      v16bf b = *(const v16bf*)(wrow[j] + kb + 16 * hi);   // B[k][n]=W[n][k], contiguous in k
#pragma unroll
      for (int i = 0; i < 4; ++i)
        acc[i][j] = __builtin_amdgcn_wmma_f32_16x16x32_bf16(false, a[i], false, b, (short)0,
                                                            acc[i][j], false, false);
    }
  }

#pragma unroll
  for (int i = 0; i < 4; ++i)
#pragma unroll
    for (int j = 0; j < 4; ++j) {
#pragma unroll
      for (int r = 0; r < 8; ++r) {
        const int row = m0 + i * 16 + r + 8 * hi;   // D: M = r + 8*(lane>=16)
        const int col = n0 + j * 16 + lm;           // N = lane & 15
        if (F32OUT) ((float*)Cout)[(size_t)row * N + col] = acc[i][j][r];
        else        ((__bf16*)Cout)[(size_t)row * N + col] = (__bf16)acc[i][j][r];
      }
    }
}

// ---------------------------------------------------------------- windowed flash attention
// 1 wave per (b,h,16-query tile). grid=(S/16, H, B), block=32.
__global__ void __launch_bounds__(32)
attn_kernel(const __bf16* __restrict__ Qr, const __bf16* __restrict__ Kr,
            const __bf16* __restrict__ Vt, __bf16* __restrict__ O,
            const int* __restrict__ winp) {
  __shared__ __align__(32) float  sS[16 * SMAXK];
  __shared__ __align__(32) __bf16 sP[16 * SMAXK];
  const int lane = threadIdx.x & 31;
  const int lm = lane & 15, hi = lane >> 4;
  const int qbase = blockIdx.x * 16;
  const int h = blockIdx.y, b = blockIdx.z;
  int W = winp[0];
  if (W > 256) W = 256;               // defensive: keeps LDS span <= SMAXK
  if (W < 1)   W = 64;
  int lo = qbase - W + 1; if (lo < 0) lo = 0;
  lo &= ~31;                          // align key-window start to 32
  int nk = (qbase + 16 - lo + 31) & ~31;

  const size_t bh = (size_t)b * H_ + h;
  const __bf16* Qp = Qr + bh * S_ * D_;
  const __bf16* Kp = Kr + bh * S_ * D_;
  const __bf16* Vp = Vt + bh * D_ * S_;

  // Q fragments over head dim (4 x K=32)
  v16bf qa[4];
  {
    const __bf16* qrow = Qp + (size_t)(qbase + lm) * D_;
#pragma unroll
    for (int t = 0; t < 4; ++t) qa[t] = load_a_frag(qrow, t * 32, hi);
  }

  const float scale = 0.08838834764831845f;   // 1/sqrt(128)

  // ---- pass 1: scores -> LDS (masked) ----
  for (int kt = 0; kt < nk; kt += 16) {
    const int key  = lo + kt + lm;
    const int keyc = key < S_ ? key : S_ - 1;     // clamp OOB reads (masked anyway)
    const __bf16* krow = Kp + (size_t)keyc * D_;
    v8f acc = {};
#pragma unroll
    for (int t = 0; t < 4; ++t) {
      v16bf bfrag = *(const v16bf*)(krow + t * 32 + 16 * hi);   // B[d][key]=K[key][d]
      acc = __builtin_amdgcn_wmma_f32_16x16x32_bf16(false, qa[t], false, bfrag, (short)0, acc, false, false);
    }
#pragma unroll
    for (int r = 0; r < 8; ++r) {
      const int q = qbase + r + 8 * hi;
      const bool keep = (key <= q) && (q - key < W);
      sS[(r + 8 * hi) * SMAXK + kt + lm] = keep ? acc[r] * scale : -3.0e38f;
    }
  }
  __syncthreads();

  // ---- softmax: 2 lanes per row (row = lane&15, halves combined via shfl_xor 16) ----
  {
    const int row = lm;
    const int c0 = hi * (nk >> 1), c1 = c0 + (nk >> 1);
    float mx = -3.0e38f;
    for (int c = c0; c < c1; ++c) mx = fmaxf(mx, sS[row * SMAXK + c]);
    mx = fmaxf(mx, __shfl_xor(mx, 16));
    float sum = 0.f;
    for (int c = c0; c < c1; ++c) {
      float p = __expf(sS[row * SMAXK + c] - mx);
      sS[row * SMAXK + c] = p;
      sum += p;
    }
    sum += __shfl_xor(sum, 16);
    const float inv = 1.0f / sum;
    for (int c = c0; c < c1; ++c) sP[row * SMAXK + c] = (__bf16)(sS[row * SMAXK + c] * inv);
  }
  __syncthreads();

  // ---- pass 2: O = P @ V (V transposed: contiguous keys per d-row) ----
  v8f o[8];
#pragma unroll
  for (int n = 0; n < 8; ++n) o[n] = v8f{};
  for (int ks = 0; ks < nk; ks += 32) {
    v16bf pa = load_a_frag(sP + lm * SMAXK, ks, hi);
#pragma unroll
    for (int n = 0; n < 8; ++n) {
      int keyb = lo + ks + 16 * hi;
      if (keyb > S_ - 16) keyb = S_ - 16;         // clamp (probs there are 0)
      v16bf vb = *(const v16bf*)(Vp + (size_t)(n * 16 + lm) * S_ + keyb);
      o[n] = __builtin_amdgcn_wmma_f32_16x16x32_bf16(false, pa, false, vb, (short)0, o[n], false, false);
    }
  }

  // ---- epilogue: O[b, q, h*128 + d] ----
  const size_t obase = ((size_t)b * S_ + qbase) * E_ + (size_t)h * D_;
#pragma unroll
  for (int n = 0; n < 8; ++n)
#pragma unroll
    for (int r = 0; r < 8; ++r)
      O[obase + (size_t)(r + 8 * hi) * E_ + n * 16 + lm] = (__bf16)o[n][r];
}

// ---------------------------------------------------------------- host launcher
extern "C" void kernel_launch(void* const* d_in, const int* in_sizes, int n_in,
                              void* d_out, int out_size, void* d_ws, size_t ws_size,
                              hipStream_t stream) {
  (void)in_sizes; (void)n_in; (void)out_size; (void)ws_size;
  const float* x     = (const float*)d_in[0];
  const float* w_qkv = (const float*)d_in[1];
  const float* w_out = (const float*)d_in[2];
  const float* w_c1  = (const float*)d_in[3];
  const float* w_c2  = (const float*)d_in[4];
  float* out = (float*)d_out;

  char* ws = (char*)d_ws;
  float*  stats  = (float*)(ws + OFF_STATS);
  int*    winp   = (int*)(stats + 4100);
  __bf16* xbf    = (__bf16*)(ws + OFF_XBF);
  __bf16* wqkvbf = (__bf16*)(ws + OFF_WQKV);
  __bf16* woutbf = (__bf16*)(ws + OFF_WOUT);
  __bf16* qkvbf  = (__bf16*)(ws + OFF_QKV);
  __bf16* Qrb    = (__bf16*)(ws + OFF_QR);
  __bf16* Krb    = (__bf16*)(ws + OFF_KR);
  __bf16* Vtb    = (__bf16*)(ws + OFF_VT);
  __bf16* Obf    = (__bf16*)(ws + OFF_O);

  const size_t nx  = (size_t)B_ * S_ * E_;         // 8388608
  const size_t nwq = (size_t)3 * E_ * E_;          // 12582912
  const size_t nwo = (size_t)E_ * E_;              // 4194304

  init_stats_kernel<<<1, 64, 0, stream>>>(stats);
  f32_to_bf16_kernel<<<(int)((nx  + 255) / 256), 256, 0, stream>>>(x, xbf, nx);
  f32_to_bf16_kernel<<<(int)((nwq + 255) / 256), 256, 0, stream>>>(w_qkv, wqkvbf, nwq);
  f32_to_bf16_kernel<<<(int)((nwo + 255) / 256), 256, 0, stream>>>(w_out, woutbf, nwo);

  colstats_kernel<<<dim3(E_, B_), 256, 0, stream>>>(x, stats);
  window_kernel<<<1, 512, 0, stream>>>(stats, w_c1, w_c2, winp);

  // qkv = x @ w_qkv^T : M=4096, N=6144, K=2048
  gemm_bf16_nt_kernel<false><<<dim3(6144 / 128, 4096 / 128), 128, 0, stream>>>(xbf, wqkvbf, qkvbf, 3 * E_, E_);

  rope_scatter_kernel<<<dim3(S_, H_, B_), D_, 0, stream>>>(qkvbf, Qrb, Krb, Vtb);

  attn_kernel<<<dim3(S_ / 16, H_, B_), 32, 0, stream>>>(Qrb, Krb, Vtb, Obf, winp);

  // out = O @ w_out^T : M=4096, N=2048, K=2048 -> f32 d_out
  gemm_bf16_nt_kernel<true><<<dim3(2048 / 128, 4096 / 128), 128, 0, stream>>>(Obf, woutbf, (void*)out, E_, E_);
}
